// ReCoLoss_50173807952175
// MI455X (gfx1250) — matrix operand order repaired and checked by reading, not compiled, not run on hardware
//
#include <hip/hip_runtime.h>

typedef __attribute__((ext_vector_type(16))) _Float16 v16h;
typedef __attribute__((ext_vector_type(8)))  float    v8f;

#define NB    2
#define NS    21
#define HW    65536          // 256*256
#define NC    256
#define NPIX  131072         // NB*HW
#define NQ    256
#define EPS_F 1e-8f

// ---------------- workspace layout (bytes) ----------------
#define OFF_PROTO    0          // float[21*256]        21504
#define OFF_COUNTS   21504      // int[32]
#define OFF_HCOUNTS  21632      // int[32]
#define OFF_HCUR     21760      // int[32]
#define OFF_SCUR     21888      // int[32]
#define OFF_ICUR     22016      // int[32]
#define OFF_OFFS     22144      // int[32]   (22 used)
#define OFF_PNORM    22272      // float[32]
#define OFF_CDF      22400      // float[448] (21*20 used)
#define OFF_SEGLOSS  24192      // float[32]
#define CTRL_BYTES   24576
#define OFF_ORDER    24576      // int[NPIX]            524288
#define OFF_FNORM    548864     // float[NPIX]          524288
#define OFF_PROTOH   1073152    // _Float16[21*256]     (pad to 16384)
#define OFF_FEATSH   1089536    // _Float16[NPIX*256]   67108864

union V16 { v16h h; uint4 u[2]; };

__device__ __forceinline__ unsigned hash3(unsigned a, unsigned b, unsigned c) {
  unsigned long long x = ((unsigned long long)a << 33) ^ ((unsigned long long)b << 11) ^ c;
  x ^= x >> 30; x *= 0xBF58476D1CE4E5B9ull;
  x ^= x >> 27; x *= 0x94D049BB133111EBull;
  x ^= x >> 31;
  return (unsigned)x;
}
__device__ __forceinline__ float u01(unsigned h) { return (float)(h >> 8) * (1.0f / 16777216.0f); }

// ------------- kernel 1: per-segment counts -------------
__global__ void count_kernel(const int* __restrict__ label, const float* __restrict__ mask,
                             const float* __restrict__ prob,
                             int* __restrict__ counts, int* __restrict__ hcounts) {
  int pix = blockIdx.x * 256 + threadIdx.x;
  int lab = label[pix];
  bool ign = lab > NS - 1; if (ign) lab = 0;
  float mm = ign ? 0.0f : mask[pix];
  bool valid = mm > 0.0f;
  int b = pix >> 16, hw = pix & 65535;
  float p = prob[((size_t)b * NS + lab) * HW + hw];
  bool hard = valid && (p < 1.0f);
  if (valid) {
    atomicAdd(&counts[lab], 1);
    if (hard) atomicAdd(&hcounts[lab], 1);
  }
}

// ------------- kernel 2: prototype sums -------------
__global__ void proto_kernel(const float* __restrict__ rep, const int* __restrict__ label,
                             const float* __restrict__ mask, float* __restrict__ proto) {
  __shared__ float ssum[NS];
  int tid = threadIdx.x;
  int c = blockIdx.x;
  if (tid < NS) ssum[tid] = 0.0f;
  __syncthreads();
  int base = blockIdx.y * 2048;
  for (int it = 0; it < 8; ++it) {
    int pix = base + it * 256 + tid;
    int lab = label[pix];
    bool ign = lab > NS - 1; if (ign) lab = 0;
    float mm = ign ? 0.0f : mask[pix];
    if (mm > 0.0f) {
      int b = pix >> 16, hw = pix & 65535;
      float v = rep[((size_t)b * NC + c) * HW + hw];
      atomicAdd(&ssum[lab], v);
    }
  }
  __syncthreads();
  if (tid < NS) atomicAdd(&proto[tid * NC + c], ssum[tid]);
}

// ------------- kernel 3: prefix offsets -------------
__global__ void offsets_kernel(const int* __restrict__ counts, int* __restrict__ offs) {
  if (threadIdx.x == 0) {
    int run = 0;
    for (int s = 0; s < NS; ++s) { offs[s] = run; run += counts[s]; }
    offs[NS] = run;
  }
}

// ------------- kernel 4: finalize protos, WMMA Gram, CDF -------------
__global__ void finalize_kernel(const int* __restrict__ counts, float* __restrict__ proto,
                                _Float16* __restrict__ phG, float* __restrict__ pnormG,
                                float* __restrict__ cdf) {
  __shared__ float red[256];
  __shared__ __align__(16) _Float16 ph[32 * NC];   // rows 21..31 zero padding
  __shared__ float pn[NS];
  __shared__ float gram[32 * 32];
  int tid = threadIdx.x;
  // zero-pad rows 21..31 so fragment loads need no predication
  for (int r = NS; r < 32; ++r) ph[r * NC + tid] = (_Float16)0.0f;
  for (int s = 0; s < NS; ++s) {
    int cnt = counts[s] > 1 ? counts[s] : 1;
    float v = proto[s * NC + tid] / (float)cnt;
    _Float16 hv = (_Float16)v;
    ph[s * NC + tid] = hv;
    phG[s * NC + tid] = hv;
    red[tid] = v * v;
    __syncthreads();
    for (int o = 128; o > 0; o >>= 1) {
      if (tid < o) red[tid] += red[tid + o];
      __syncthreads();
    }
    if (tid == 0) { float nn = fmaxf(sqrtf(red[0]), EPS_F); pn[s] = nn; pnormG[s] = nn; }
    __syncthreads();
  }
  // Gram = P * P^T via v_wmma_f32_16x16x32_f16 (wave 0, EXEC all ones)
  if (tid < 32) {
    int lane = tid;
#pragma unroll
    for (int mt = 0; mt < 2; ++mt)
#pragma unroll
      for (int nt = 0; nt < 2; ++nt) {
        v8f c = {0.f, 0.f, 0.f, 0.f, 0.f, 0.f, 0.f, 0.f};
        const _Float16* arow = ph + (size_t)(mt * 16 + (lane & 15)) * NC;
        const _Float16* brow = ph + (size_t)(nt * 16 + (lane & 15)) * NC;
#pragma unroll
        for (int kb = 0; kb < 8; ++kb) {
          V16 av, bv;
          int p0 = kb * 32 + ((lane & 16) ? 8 : 0);
          av.u[0] = *(const uint4*)(arow + p0);
          av.u[1] = *(const uint4*)(arow + p0 + 16);
          int q0 = kb * 32 + ((lane & 16) ? 8 : 0);
          bv.u[0] = *(const uint4*)(brow + q0);
          bv.u[1] = *(const uint4*)(brow + q0 + 16);
          c = __builtin_amdgcn_wmma_f32_16x16x32_f16(false, av.h, false, bv.h,
                                                     (short)0, c, false, false);
        }
#pragma unroll
        for (int r = 0; r < 8; ++r) {
          int M = mt * 16 + r + ((lane & 16) ? 8 : 0);
          int Nn = nt * 16 + (lane & 15);
          gram[M * 32 + Nn] = c[r];
        }
      }
  }
  __syncthreads();
  if (tid < NS) {
    float ni = pn[tid];
    float lg[20];
    float mx = -1e30f;
    for (int k = 0; k < 20; ++k) {
      int j = tid + 1 + k; if (j >= NS) j -= NS;
      float cs = gram[tid * 32 + j] / (ni * pn[j]);
      lg[k] = cs * 2.0f;  // / TEMP(0.5)
      mx = fmaxf(mx, lg[k]);
    }
    float z = 0.0f;
    for (int k = 0; k < 20; ++k) z += __expf(lg[k] - mx);
    float run = 0.0f;
    for (int k = 0; k < 20; ++k) { run += __expf(lg[k] - mx) / z; cdf[tid * 20 + k] = run; }
  }
}

// ------------- kernel 5: fp32 -> f16 features + norms (LDS transpose) -------------
__global__ void convert_kernel(const float* __restrict__ rep, _Float16* __restrict__ fh,
                               float* __restrict__ fnorm) {
  __shared__ float tile[NC][33];
  __shared__ float nacc[32];
  int tid = threadIdx.x;
  int pix0 = blockIdx.x * 32;
  int b = pix0 >> 16;
  int hw0 = pix0 & 65535;
  int hwl = tid & 31;
  int c8 = tid >> 5;
  if (tid < 32) nacc[tid] = 0.0f;
  __syncthreads();
  float psum = 0.0f;
  for (int cc = 0; cc < 32; ++cc) {
    int c = cc * 8 + c8;
    float v = rep[((size_t)b * NC + c) * HW + hw0 + hwl];
    tile[c][hwl] = v;
    psum += v * v;
  }
  atomicAdd(&nacc[hwl], psum);
  __syncthreads();
  if (tid < 32) fnorm[pix0 + tid] = sqrtf(nacc[tid]);
  for (int p = 0; p < 32; ++p)
    fh[(size_t)(pix0 + p) * NC + tid] = (_Float16)tile[tid][p];
}

// ------------- kernel 6: hard-first scatter into segment blocks -------------
__global__ void scatter_kernel(const int* __restrict__ label, const float* __restrict__ mask,
                               const float* __restrict__ prob, const int* __restrict__ hcounts,
                               const int* __restrict__ offs, int* __restrict__ hcur,
                               int* __restrict__ scur, int* __restrict__ icur,
                               int* __restrict__ order) {
  int pix = blockIdx.x * 256 + threadIdx.x;
  int lab = label[pix];
  bool ign = lab > NS - 1; if (ign) lab = 0;
  float mm = ign ? 0.0f : mask[pix];
  bool valid = mm > 0.0f;
  int b = pix >> 16, hw = pix & 65535;
  float p = prob[((size_t)b * NS + lab) * HW + hw];
  bool hard = valid && (p < 1.0f);
  int pos;
  if (!valid)    pos = offs[NS] + atomicAdd(icur, 1);
  else if (hard) pos = offs[lab] + atomicAdd(&hcur[lab], 1);
  else           pos = offs[lab] + hcounts[lab] + atomicAdd(&scur[lab], 1);
  order[pos] = pix;
}

// ------------- kernel 7: contrastive loss (WMMA GEMV, 1 query/wave) -------------
__global__ void loss_kernel(const _Float16* __restrict__ fh, const _Float16* __restrict__ ph,
                            const float* __restrict__ fnorm, const float* __restrict__ pnormG,
                            const float* __restrict__ cdf, const int* __restrict__ counts,
                            const int* __restrict__ hcounts, const int* __restrict__ offs,
                            const int* __restrict__ order, float* __restrict__ segloss) {
  __shared__ __align__(16) _Float16 sA[8][NC];
  int tid = threadIdx.x;
  int wv = tid >> 5, lane = tid & 31;
  int qg = blockIdx.x * 8 + wv;      // 0..5375
  int seg = qg >> 8;
  // --- anchor: random hard pixel of segment ---
  int hc = hcounts[seg] > 1 ? hcounts[seg] : 1;
  unsigned ha = hash3(0x243F6A88u, (unsigned)qg, 0u);
  int hidx = offs[seg] + (int)(ha % (unsigned)hc);
  if (hidx > NPIX - 1) hidx = NPIX - 1;
  int apix = order[hidx];
  float anorm = fmaxf(fnorm[apix], EPS_F);
  {  // stage anchor row (512B) into this wave's LDS slot
    const uint4* src = (const uint4*)(fh + (size_t)apix * NC);
    ((uint4*)&sA[wv][0])[lane] = src[lane];
  }
  __syncthreads();

  const _Float16* anc = &sA[wv][0];
  int n = lane & 15;
  float m = -1e30f, se = 0.0f, l0 = 0.0f;
  float invA = 1.0f / anorm;

  for (int g = 0; g < 17; ++g) {
    int s = g * 16 + n;                 // slot: 0 = proto, 1..256 = negatives
    const _Float16* rowp;
    float bnorm;
    bool slot_ok = (s <= 256);
    if (s == 0) {
      rowp = ph + (size_t)seg * NC;
      bnorm = pnormG[seg];
    } else if (slot_ok) {
      unsigned h1 = hash3(0x85A308D3u, (unsigned)qg, (unsigned)s);
      float u1 = u01(h1);
      int k = 19;
      for (int t = 0; t < 20; ++t) { if (u1 < cdf[seg * 20 + t]) { k = t; break; } }
      int sn = seg + 1 + k; if (sn >= NS) sn -= NS;
      unsigned h2 = hash3(0x13198A2Eu, (unsigned)qg, (unsigned)s);
      float u2 = u01(h2);
      int cn = counts[sn] > 1 ? counts[sn] : 1;
      int nix = offs[sn] + (int)(u2 * (float)cn);
      if (nix > NPIX - 1) nix = NPIX - 1;
      int pix = order[nix];
      rowp = fh + (size_t)pix * NC;
      bnorm = fmaxf(fnorm[pix], EPS_F);
    } else {
      rowp = ph + (size_t)seg * NC;     // dummy, masked below
      bnorm = 1.0f;
    }
    // D = A(anchor replicated 16 rows) x B(16 gathered rows as columns), K=256
    v8f c = {0.f, 0.f, 0.f, 0.f, 0.f, 0.f, 0.f, 0.f};
#pragma unroll
    for (int kb = 0; kb < 8; ++kb) {
      V16 av, bv;
      int p0 = kb * 32 + ((lane & 16) ? 8 : 0);
      av.u[0] = *(const uint4*)(anc + p0);
      av.u[1] = *(const uint4*)(anc + p0 + 16);
      int q0 = kb * 32 + ((lane & 16) ? 8 : 0);
      bv.u[0] = *(const uint4*)(rowp + q0);
      bv.u[1] = *(const uint4*)(rowp + q0 + 16);
      c = __builtin_amdgcn_wmma_f32_16x16x32_f16(false, av.h, false, bv.h,
                                                 (short)0, c, false, false);
    }
    float dot = c[0];                                 // row M==0 (all rows identical)
    float lg = slot_ok ? (dot * invA / bnorm) * 2.0f  // cos / TEMP
                       : -1e30f;
    if (g == 0) l0 = __shfl(lg, 0, 32);
    float nm = fmaxf(m, lg);
    se = se * __expf(m - nm) + __expf(lg - nm);
    m = nm;
  }
  // reduce logsumexp across 16 lanes (halves hold duplicates)
  for (int o = 8; o >= 1; o >>= 1) {
    float mo = __shfl_xor(m, o, 32);
    float so = __shfl_xor(se, o, 32);
    float nm = fmaxf(m, mo);
    se = se * __expf(m - nm) + so * __expf(mo - nm);
    m = nm;
  }
  float ce = (m + __logf(se)) - l0;
  if (lane == 0 && hcounts[seg] > 0)
    atomicAdd(&segloss[seg], ce * (1.0f / (float)NQ));
}

// ------------- kernel 8: final reduce -------------
__global__ void final_kernel(const float* __restrict__ segloss, const int* __restrict__ counts,
                             float* __restrict__ out) {
  if (threadIdx.x == 0) {
    float tot = 0.0f; int vs = 0;
    for (int s = 0; s < NS; ++s) { tot += segloss[s]; if (counts[s] > 0) vs++; }
    int d = vs > 1 ? vs : 1;
    out[0] = (vs > 1) ? tot / (float)d : 0.0f;
  }
}

extern "C" void kernel_launch(void* const* d_in, const int* in_sizes, int n_in,
                              void* d_out, int out_size, void* d_ws, size_t ws_size,
                              hipStream_t stream) {
  (void)in_sizes; (void)n_in; (void)out_size; (void)ws_size;
  const float* rep   = (const float*)d_in[0];
  const int*   label = (const int*)d_in[1];
  const float* mask  = (const float*)d_in[2];
  const float* prob  = (const float*)d_in[3];
  float* out = (float*)d_out;

  char* w = (char*)d_ws;
  float*    proto   = (float*)(w + OFF_PROTO);
  int*      counts  = (int*)(w + OFF_COUNTS);
  int*      hcounts = (int*)(w + OFF_HCOUNTS);
  int*      hcur    = (int*)(w + OFF_HCUR);
  int*      scur    = (int*)(w + OFF_SCUR);
  int*      icur    = (int*)(w + OFF_ICUR);
  int*      offs    = (int*)(w + OFF_OFFS);
  float*    pnorm   = (float*)(w + OFF_PNORM);
  float*    cdf     = (float*)(w + OFF_CDF);
  float*    segloss = (float*)(w + OFF_SEGLOSS);
  int*      order   = (int*)(w + OFF_ORDER);
  float*    fnorm   = (float*)(w + OFF_FNORM);
  _Float16* protoH  = (_Float16*)(w + OFF_PROTOH);
  _Float16* featsH  = (_Float16*)(w + OFF_FEATSH);

  hipMemsetAsync(w, 0, CTRL_BYTES, stream);
  count_kernel<<<NPIX / 256, 256, 0, stream>>>(label, mask, prob, counts, hcounts);
  proto_kernel<<<dim3(NC, NPIX / 2048), 256, 0, stream>>>(rep, label, mask, proto);
  offsets_kernel<<<1, 32, 0, stream>>>(counts, offs);
  finalize_kernel<<<1, 256, 0, stream>>>(counts, proto, protoH, pnorm, cdf);
  convert_kernel<<<NPIX / 32, 256, 0, stream>>>(rep, featsH, fnorm);
  scatter_kernel<<<NPIX / 256, 256, 0, stream>>>(label, mask, prob, hcounts, offs,
                                                 hcur, scur, icur, order);
  loss_kernel<<<(NS * NQ) / 8, 256, 0, stream>>>(featsH, protoH, fnorm, pnorm, cdf,
                                                 counts, hcounts, offs, order, segloss);
  final_kernel<<<1, 32, 0, stream>>>(segloss, counts, out);
}
